// Attention_2181843386390
// MI455X (gfx1250) — compile-verified, matrix operand order checked by reading, not run on hardware
//
#include <hip/hip_runtime.h>
#include <cmath>

typedef __attribute__((ext_vector_type(16))) _Float16 v16h;
typedef __attribute__((ext_vector_type(8)))  float    v8f;

namespace {
constexpr int BATCH = 2;
constexpr int HEADS = 16;
constexpr int SEQ   = 2048;
constexpr int DH    = 64;
constexpr int BQ    = 64;                 // q rows per workgroup (4 waves x 16)
constexpr int BK    = 64;                 // keys staged per iteration
constexpr int WAVES = 4;
constexpr int NTHR  = WAVES * 32;         // 128 threads, 4 wave32
constexpr int STG   = (BK * DH / 4) / NTHR;      // float4 per thread per matrix = 8
constexpr int MSTG  = (BQ * BK / 4) / NTHR;      // int4 chunks per thread for mask = 8
constexpr float SCALE = 0.125f;           // 64^-0.5
}

// reductions across the 16 lanes of each wave32 half (C-fragment N dimension)
__device__ __forceinline__ float row_max16(float v) {
  v = fmaxf(v, __shfl_xor(v, 1, 32));
  v = fmaxf(v, __shfl_xor(v, 2, 32));
  v = fmaxf(v, __shfl_xor(v, 4, 32));
  v = fmaxf(v, __shfl_xor(v, 8, 32));
  return v;
}
__device__ __forceinline__ float row_sum16(float v) {
  v += __shfl_xor(v, 1, 32);
  v += __shfl_xor(v, 2, 32);
  v += __shfl_xor(v, 4, 32);
  v += __shfl_xor(v, 8, 32);
  return v;
}

__global__ __launch_bounds__(NTHR, 1)
void flash_attn_f16_wmma(const float* __restrict__ Q,
                         const float* __restrict__ K,
                         const float* __restrict__ V,
                         const int*   __restrict__ MASK,
                         float* __restrict__ OUT)
{
  __shared__ _Float16 Ks[BK][DH];          // key block, f16
  __shared__ _Float16 Vs[BK][DH];          // value block, f16
  __shared__ _Float16 Ps[WAVES][16][BK];   // per-wave P re-layout scratch
  __shared__ int      Ms[BQ][BK];          // mask block (async-staged)

  const int tid  = threadIdx.x;
  const int lane = tid & 31;
  const int wave = tid >> 5;

  const int qtiles = SEQ / BQ;
  const int bh = blockIdx.x / qtiles;
  const int qt = blockIdx.x % qtiles;
  const int b  = bh / HEADS;

  const float* qb = Q + (size_t)bh * SEQ * DH;
  const float* kb = K + (size_t)bh * SEQ * DH;
  const float* vb = V + (size_t)bh * SEQ * DH;
  const int*   mb = MASK + (size_t)b * SEQ * SEQ;
  float*       ob = OUT + (size_t)bh * SEQ * DH;

  const int m16   = lane & 15;
  const int half8 = (lane >> 4) << 3;      // 0 for lanes 0..15, 8 for 16..31
  const int qblk  = qt * BQ;
  const int q0    = qblk + wave * 16;      // this wave's first q row

  // ---- Q A-fragments with softmax scale folded in (exact *0.125 in f16)
  // ISA A layout: lanes 0-15 hold K in {0..7,16..23}; lanes 16-31 hold {8..15,24..31}
  v16h qa[2];
  #pragma unroll
  for (int c = 0; c < 2; ++c)
    #pragma unroll
    for (int j = 0; j < 16; ++j) {
      const int kk = c * 32 + half8 + (j & 7) + ((j >> 3) << 4);
      qa[c][j] = (_Float16)(qb[(size_t)(q0 + m16) * DH + kk] * SCALE);
    }

  v8f oacc[4] = {};          // output accumulators: 4 Dh-tiles of 16
  float mrun[8], lrun[8];    // running row max / sum (8 rows per half-wave)
  #pragma unroll
  for (int r = 0; r < 8; ++r) { mrun[r] = -INFINITY; lrun[r] = 0.f; }

  // ---- software pipeline: preload K/V block 0 into registers
  float4 kreg[STG], vreg[STG];
  {
    const float4* ks = (const float4*)kb;
    const float4* vs = (const float4*)vb;
    #pragma unroll
    for (int it = 0; it < STG; ++it) {
      const int i = it * NTHR + tid;
      kreg[it] = ks[i];
      vreg[it] = vs[i];
    }
  }

  const uint32_t ms_lds = (uint32_t)(uintptr_t)(&Ms[0][0]);  // low 32 bits = LDS offset

  for (int k0 = 0; k0 < SEQ; k0 += BK) {
    __syncthreads();   // all waves done reading previous LDS contents

    // ---- async mask block -> LDS (per-lane 16B, tracked by ASYNCcnt)
    #pragma unroll
    for (int it = 0; it < MSTG; ++it) {
      const int i   = it * NTHR + tid;
      const int row = i >> 4;          // / (BK/4)
      const int col = (i & 15) << 2;   // *4 ints
      const unsigned long long ga = (unsigned long long)(uintptr_t)
          (mb + (size_t)(qblk + row) * SEQ + (k0 + col));
      const uint32_t la = ms_lds + (uint32_t)(i << 4);
      asm volatile("global_load_async_to_lds_b128 %0, %1, off"
                   :: "v"(la), "v"(ga) : "memory");
    }

    // ---- commit pipelined K/V registers to LDS as f16
    #pragma unroll
    for (int it = 0; it < STG; ++it) {
      const int i   = it * NTHR + tid;
      const int row = i >> 4;
      const int col = (i & 15) << 2;
      Ks[row][col + 0] = (_Float16)kreg[it].x;
      Ks[row][col + 1] = (_Float16)kreg[it].y;
      Ks[row][col + 2] = (_Float16)kreg[it].z;
      Ks[row][col + 3] = (_Float16)kreg[it].w;
      Vs[row][col + 0] = (_Float16)vreg[it].x;
      Vs[row][col + 1] = (_Float16)vreg[it].y;
      Vs[row][col + 2] = (_Float16)vreg[it].z;
      Vs[row][col + 3] = (_Float16)vreg[it].w;
    }
    // ---- issue next block's global loads now; latency hides under WMMA work
    if (k0 + BK < SEQ) {
      const float4* ks = (const float4*)(kb + (size_t)(k0 + BK) * DH);
      const float4* vs = (const float4*)(vb + (size_t)(k0 + BK) * DH);
      #pragma unroll
      for (int it = 0; it < STG; ++it) {
        const int i = it * NTHR + tid;
        kreg[it] = ks[i];
        vreg[it] = vs[i];
      }
    }

    asm volatile("s_wait_asynccnt 0x0" ::: "memory");  // mask block landed in LDS
    __syncthreads();

    // ---- scores: S = (Q*scale) * K^T, 4 key tiles of 16, K-dim 64 (2 WMMA each)
    v8f sc[4];
    #pragma unroll
    for (int t = 0; t < 4; ++t) {
      v8f acc = {};
      #pragma unroll
      for (int c = 0; c < 2; ++c) {
        v16h bf;   // lane holds key column t*16+m16, j walks Dh
        #pragma unroll
        for (int j = 0; j < 16; ++j) {
          const int kk = c * 32 + half8 + (j & 7) + ((j >> 3) << 4);
          bf[j] = Ks[t * 16 + m16][kk];
        }
        acc = __builtin_amdgcn_wmma_f32_16x16x32_f16(false, qa[c], false, bf,
                                                     (short)0, acc, false, false);
      }
      sc[t] = acc;
    }

    // ---- mask from LDS (mask==0 -> -1e9, matching reference)
    #pragma unroll
    for (int t = 0; t < 4; ++t)
      #pragma unroll
      for (int r = 0; r < 8; ++r) {
        if (Ms[wave * 16 + r + half8][t * 16 + m16] == 0) sc[t][r] = -1e9f;
      }

    // ---- online softmax update
    float mnew[8], corr[8], bsum[8];
    #pragma unroll
    for (int r = 0; r < 8; ++r) {
      float bm = fmaxf(fmaxf(sc[0][r], sc[1][r]), fmaxf(sc[2][r], sc[3][r]));
      bm = row_max16(bm);
      mnew[r] = fmaxf(mrun[r], bm);
      corr[r] = __expf(mrun[r] - mnew[r]);   // exp(-inf)=0 on first block
      bsum[r] = 0.f;
    }
    #pragma unroll
    for (int t = 0; t < 4; ++t)
      #pragma unroll
      for (int r = 0; r < 8; ++r) {
        const float p = __expf(sc[t][r] - mnew[r]);
        bsum[r] += p;
        Ps[wave][r + half8][t * 16 + m16] = (_Float16)p;  // C-layout -> LDS
      }
    #pragma unroll
    for (int r = 0; r < 8; ++r) {
      lrun[r] = lrun[r] * corr[r] + row_sum16(bsum[r]);
      mrun[r] = mnew[r];
    }
    #pragma unroll
    for (int t = 0; t < 4; ++t)
      #pragma unroll
      for (int r = 0; r < 8; ++r)
        oacc[t][r] *= corr[r];

    // wave-local LDS RAW fence for Ps (write in C-layout, read in A-layout)
    asm volatile("s_wait_dscnt 0x0" ::: "memory");

    // ---- O += P * V   (A = P 16x64 over 2 chunks, B = V 32x16 tiles)
    v16h pa[2];
    #pragma unroll
    for (int c = 0; c < 2; ++c)
      #pragma unroll
      for (int j = 0; j < 16; ++j) {
        const int kk = c * 32 + half8 + (j & 7) + ((j >> 3) << 4);
        pa[c][j] = Ps[wave][m16][kk];
      }
    #pragma unroll
    for (int t = 0; t < 4; ++t) {
      #pragma unroll
      for (int c = 0; c < 2; ++c) {
        v16h bv;   // lane holds Dh column t*16+m16, j walks key dim
        #pragma unroll
        for (int j = 0; j < 16; ++j) {
          const int kk = c * 32 + half8 + (j & 7) + ((j >> 3) << 4);
          bv[j] = Vs[kk][t * 16 + m16];
        }
        oacc[t] = __builtin_amdgcn_wmma_f32_16x16x32_f16(false, pa[c], false, bv,
                                                         (short)0, oacc[t], false, false);
      }
    }
  }

  // ---- normalize by row sum and store fp32
  #pragma unroll
  for (int t = 0; t < 4; ++t)
    #pragma unroll
    for (int r = 0; r < 8; ++r) {
      const int qrow = q0 + r + half8;
      ob[(size_t)qrow * DH + t * 16 + m16] = oacc[t][r] / lrun[r];
    }
}

extern "C" void kernel_launch(void* const* d_in, const int* in_sizes, int n_in,
                              void* d_out, int out_size, void* d_ws, size_t ws_size,
                              hipStream_t stream) {
  const float* q    = (const float*)d_in[0];
  const float* k    = (const float*)d_in[1];
  const float* v    = (const float*)d_in[2];
  const int*   mask = (const int*)d_in[3];
  float*       out  = (float*)d_out;

  dim3 grid(BATCH * HEADS * (SEQ / BQ));   // 2*16*32 = 1024 workgroups
  dim3 block(NTHR);                        // 128 threads = 4 waves (wave32)
  flash_attn_f16_wmma<<<grid, block, 0, stream>>>(q, k, v, mask, out);
}
